// TemporalRefinementNetwork_86260123173414
// MI455X (gfx1250) — matrix operand order repaired and checked by reading, not compiled, not run on hardware
//
#include <hip/hip_runtime.h>
#include <hip/hip_bf16.h>
#include <math.h>

// ---------------------------------------------------------------------------
// TemporalRefinementNetwork for MI455X (gfx1250, wave32, WMMA).
// Heavy GEMMs (in_proj, MLP1, MLP2) run on v_wmma_f32_16x16x32_bf16 with
// fp32 accumulation. Residual stream for one track stays in LDS across the
// whole 4-block stack. One workgroup (8 wave32 waves) per query track.
// Bulk LDS<->HBM staging uses the gfx1250 async-LDS path (ASYNCcnt).
// ---------------------------------------------------------------------------

typedef __attribute__((ext_vector_type(16))) __bf16 bf16x16;
typedef __attribute__((ext_vector_type(8)))  float  f32x8;

union FragBF {           // one WMMA 16-bf16 operand = 8 VGPRs = 2x uint4
    bf16x16 v;
    uint4   q[2];
};

// ---- problem constants (match reference setup_inputs) ----
#define QN   128
#define TN   64
#define CN   64
#define HN   128
#define WN   128
#define HID  128
#define KWIN 49

// ---- workspace layout (bytes, 256-aligned) ----
#define WS_TRACKS 0u               // 16384 f32
#define WS_HID    65536u           // 8192*128 f32
#define WS_XIN    4259840u         // 8192*288 bf16
#define WS_INT    8978432u         // 128*288 bf16   (in_proj^T, zero-padded K)
#define WS_W1T    9052160u         // 4*256*128 bf16 (mlp_w1^T)
#define WS_W2T    9314304u         // 4*128*256 bf16 (mlp_w2^T)

__device__ __forceinline__ float bsum32(float v) {
    #pragma unroll
    for (int o = 16; o; o >>= 1) v += __shfl_xor(v, o, 32);
    return v;
}
__device__ __forceinline__ float bmax32(float v) {
    #pragma unroll
    for (int o = 16; o; o >>= 1) v = fmaxf(v, __shfl_xor(v, o, 32));
    return v;
}
__device__ __forceinline__ float fm_at(const float* __restrict__ pl, int y, int x) {
    if (y < 0 || y > (HN - 1) || x < 0 || x > (WN - 1)) return 0.0f;
    return pl[(y << 7) + x];
}
__device__ __forceinline__ float gelu_exact(float v) {
    return 0.5f * v * (1.0f + erff(v * 0.70710678118654752f));
}
// LDS byte offset of a generic pointer into shared memory (aperture keeps the
// wave-relative LDS offset in addr[31:0] on gfx1250).
__device__ __forceinline__ unsigned lds_off(const void* p) {
    return (unsigned)(uintptr_t)p;
}

// ---------------------------------------------------------------------------
// prep: weight transpose/convert to bf16, tracks copy, hidden zero.
// ---------------------------------------------------------------------------
__global__ __launch_bounds__(256)
void prep_kernel(const float* __restrict__ coarse, const float* __restrict__ w_in,
                 const float* __restrict__ w1, const float* __restrict__ w2,
                 float* __restrict__ tracks, float* __restrict__ hid,
                 __bf16* __restrict__ inT, __bf16* __restrict__ W1T,
                 __bf16* __restrict__ W2T)
{
    int i = blockIdx.x * 256 + threadIdx.x;
    const int n0 = 16384, n1 = 8192 * 128, n2 = 128 * 288, n3 = 4 * 256 * 128;
    if (i < n0) { tracks[i] = coarse[i]; return; }
    i -= n0;
    if (i < n1) { hid[i] = 0.0f; return; }
    i -= n1;
    if (i < n2) {                                     // inT[n][k] = in_w[k][n], pad k>=276
        int n = i / 288, k = i % 288;
        inT[i] = (k < 276) ? (__bf16)w_in[k * 128 + n] : (__bf16)0.0f;
        return;
    }
    i -= n2;
    if (i < n3) {                                     // W1T[b][n][k] = w1[b][k][n]
        int b = i >> 15, r = i & 32767, n = r >> 7, k = r & 127;
        W1T[i] = (__bf16)w1[(b * 128 + k) * 256 + n];
        return;
    }
    i -= n3;
    if (i < n3) {                                     // W2T[b][n][k] = w2[b][k][n]
        int b = i >> 15, r = i & 32767, n = r >> 8, k = r & 255;
        W2T[i] = (__bf16)w2[(b * 256 + k) * 128 + n];
        return;
    }
}

// ---------------------------------------------------------------------------
// correlation + softmax delta + positional encoding + token assembly.
// One wave32 per (q,t); lane owns channels {lane, lane+32}.
// ---------------------------------------------------------------------------
__global__ __launch_bounds__(256)
void corr_kernel(const float* __restrict__ fm, const float* __restrict__ qf,
                 float* __restrict__ tracks, const float* __restrict__ hid,
                 __bf16* __restrict__ xin)
{
    const int wid  = (blockIdx.x * 256 + threadIdx.x) >> 5;   // 0..8191
    const int lane = threadIdx.x & 31;
    const int q = wid >> 6, t = wid & 63;

    const float* p0 = fm + ((size_t)t * CN + lane) * (HN * WN);
    const float* p1 = p0 + 32 * (HN * WN);
    const float qc0 = qf[q * CN + lane];
    const float qc1 = qf[q * CN + lane + 32];
    const float qnorm = fmaxf(sqrtf(bsum32(qc0 * qc0 + qc1 * qc1)), 1e-12f);

    const int ti = (q * TN + t) * 2;
    float px = tracks[ti], py = tracks[ti + 1];
    const float fx = px * 0.25f, fy = py * 0.25f;     // feature_stride = 4

    float corrA = 0.0f, corrB = 0.0f, lgA = -1e30f, lgB = -1e30f;
    #pragma unroll
    for (int p = 0; p < KWIN; ++p) {
        const float cx = fx + (float)(p % 7 - 3);
        const float cy = fy + (float)(p / 7 - 3);
        const float x0f = floorf(cx), y0f = floorf(cy);
        const float wx = cx - x0f, wy = cy - y0f;
        const int x0 = (int)x0f, y0 = (int)y0f;
        const float w00 = (1.0f - wx) * (1.0f - wy), w01 = wx * (1.0f - wy);
        const float w10 = (1.0f - wx) * wy,          w11 = wx * wy;
        const float s0 = fm_at(p0, y0, x0) * w00 + fm_at(p0, y0, x0 + 1) * w01
                       + fm_at(p0, y0 + 1, x0) * w10 + fm_at(p0, y0 + 1, x0 + 1) * w11;
        const float s1 = fm_at(p1, y0, x0) * w00 + fm_at(p1, y0, x0 + 1) * w01
                       + fm_at(p1, y0 + 1, x0) * w10 + fm_at(p1, y0 + 1, x0 + 1) * w11;
        const float pc = bsum32(s0 * qc0 + s1 * qc1);        // raw corr
        const float pn = bsum32(s0 * s0 + s1 * s1);          // |sample|^2
        const float lg = pc / (fmaxf(sqrtf(pn), 1e-12f) * qnorm);
        if (p < 32) { if (lane == p)      { corrA = pc; lgA = lg; } }
        else        { if (lane == p - 32) { corrB = pc; lgB = lg; } }
    }

    // softmax over 49 logits * (1/TEMP), lane-resident (A: p=lane, B: p=lane+32)
    const float sA = lgA * 10.0f;
    const float sB = (lane < 17) ? lgB * 10.0f : -1e30f;
    const float M  = bmax32(fmaxf(sA, sB));
    const float eA = expf(sA - M);
    const float eB = (lane < 17) ? expf(sB - M) : 0.0f;
    const float S  = bsum32(eA + eB);
    const float pA = eA / S, pB = eB / S;
    const float oxA = (float)((lane % 7) - 3) * 4.0f;
    const float oyA = (float)((lane / 7) - 3) * 4.0f;
    const int   pb  = lane + 32;
    const float oxB = (float)((pb % 7) - 3) * 4.0f;
    const float oyB = (float)((pb / 7) - 3) * 4.0f;
    px += bsum32(pA * oxA + pB * oxB);
    py += bsum32(pA * oyA + pB * oyB);
    if (lane == 0) { tracks[ti] = px; tracks[ti + 1] = py; }

    // assemble 288-wide bf16 token: corr[49] | posenc[99] | hidden[128] | pad[12]
    __bf16* xr = xin + (size_t)wid * 288;
    xr[lane] = (__bf16)corrA;
    if (lane < 17) xr[32 + lane] = (__bf16)corrB;

    const float tt = (float)t * (1.0f / 63.0f);
    for (int j = lane; j < 99; j += 32) {
        float v;
        if (j < 3) v = (j == 0) ? px : ((j == 1) ? py : tt);
        else {
            const int jj = j - 3, dim = jj >> 5, r = jj & 31, band = r & 15;
            const float freq = exp2f((float)band * 0.6f) * 3.14159265358979f; // 2^(band*9/15)*pi
            const float c = (dim == 0) ? px : ((dim == 1) ? py : tt);
            const float a = c * freq;
            v = (r >= 16) ? cosf(a) : sinf(a);
        }
        xr[49 + j] = (__bf16)v;
    }
    const float* hrow = hid + (size_t)wid * 128;
    for (int j = lane; j < 128; j += 32) xr[148 + j] = (__bf16)hrow[j];
    if (lane < 12) xr[276 + lane] = (__bf16)0.0f;
}

// ---------------------------------------------------------------------------
// LayerNorm helper: 4 lanes per token, width-4 shuffle reduction, bf16 out.
// ---------------------------------------------------------------------------
__device__ __forceinline__ void layernorm_bf16(const float* __restrict__ Xf,
                                               __bf16* __restrict__ dst,
                                               const float* __restrict__ gg,
                                               const float* __restrict__ bb, int tid)
{
    const int t = tid >> 2, sub = tid & 3;
    const float* row = Xf + t * HID;
    float s = 0.0f;
    #pragma unroll
    for (int i = 0; i < 32; ++i) s += row[sub * 32 + i];
    s += __shfl_xor(s, 1, 32); s += __shfl_xor(s, 2, 32);
    const float mean = s * (1.0f / 128.0f);
    float v = 0.0f;
    #pragma unroll
    for (int i = 0; i < 32; ++i) { const float d = row[sub * 32 + i] - mean; v += d * d; }
    v += __shfl_xor(v, 1, 32); v += __shfl_xor(v, 2, 32);
    const float rstd = rsqrtf(v * (1.0f / 128.0f) + 1e-5f);
    #pragma unroll
    for (int i = 0; i < 32; ++i) {
        const int d = sub * 32 + i;
        dst[t * HID + d] = (__bf16)((row[d] - mean) * rstd * gg[d] + bb[d]);
    }
}

// ---------------------------------------------------------------------------
// refine: one workgroup per track q. in_proj + 4 ConvNeXt-ish blocks + heads.
// LDS: Xf f32[64][128] | XB bf16[64][128] | LNX bf16[64][128] |
//      HB bf16[64][256] | XIN bf16[64][288]   (135168 bytes dynamic)
// ---------------------------------------------------------------------------
__global__ __launch_bounds__(256)
void refine_kernel(const __bf16* __restrict__ xin, const __bf16* __restrict__ inT,
                   const __bf16* __restrict__ W1T, const __bf16* __restrict__ W2T,
                   const float* __restrict__ in_b,
                   const float* __restrict__ ln1g, const float* __restrict__ ln1b,
                   const float* __restrict__ convw, const float* __restrict__ convb,
                   const float* __restrict__ ln2g, const float* __restrict__ ln2b,
                   const float* __restrict__ b1, const float* __restrict__ b2,
                   const float* __restrict__ dw, const float* __restrict__ db,
                   const float* __restrict__ vw1, const float* __restrict__ vb1,
                   const float* __restrict__ vw2, const float* __restrict__ vb2,
                   float* __restrict__ tracks, float* __restrict__ hid,
                   float* __restrict__ out_tracks, float* __restrict__ out_vis,
                   int last)
{
    extern __shared__ char smem[];
    float*  Xf  = (float*)smem;                 // residual stream (f32)
    __bf16* XB  = (__bf16*)(smem + 32768);      // LN2 out (WMMA A source)
    __bf16* LNX = (__bf16*)(smem + 49152);      // LN1 out (conv source)
    __bf16* HB  = (__bf16*)(smem + 65536);      // MLP hidden (bf16)
    __bf16* XIN = (__bf16*)(smem + 98304);      // input tokens

    const int q = blockIdx.x;
    const int tid = threadIdx.x;
    const int w = tid >> 5, lane = tid & 31;
    const int lm = lane & 15, g = lane >> 4;
    const int mt = w >> 1;                      // M-tile for this wave (0..3)

    // ---- stage input tokens into LDS via async global->LDS (ASYNCcnt) ----
    {
        const char* gsrc = (const char*)(xin + (size_t)q * TN * 288);
        const unsigned lbase = lds_off(XIN);
        for (int i = tid; i < TN * 288 * 2 / 16; i += 256) {
            const unsigned loff = lbase + i * 16;
            const char* ga = gsrc + i * 16;
            asm volatile("global_load_async_to_lds_b128 %0, %1, off"
                         :: "v"(loff), "v"(ga) : "memory");
        }
        asm volatile("s_wait_asynccnt 0x0" ::: "memory");
    }
    __syncthreads();

    // ---- in_proj: X[64,288] @ W[288,128]  (4 N-tiles per wave, 9 K-steps) ----
    {
        f32x8 acc[4] = {};
        for (int kt = 0; kt < 9; ++kt) {
            FragBF a;
            const char* ab = (const char*)XIN + (mt * 16 + lm) * 576 + kt * 64 + g * 16;
            a.q[0] = *(const uint4*)ab;
            a.q[1] = *(const uint4*)(ab + 32);
            #pragma unroll
            for (int j = 0; j < 4; ++j) {
                const int n = ((w & 1) * 4 + j) * 16 + lm;
                const __bf16* wp = inT + n * 288 + kt * 32 + g * 8;
                FragBF b;
                b.q[0] = *(const uint4*)wp;
                b.q[1] = *(const uint4*)(wp + 16);
                acc[j] = __builtin_amdgcn_wmma_f32_16x16x32_bf16(
                    false, a.v, false, b.v, (short)0, acc[j], false, false);
            }
        }
        #pragma unroll
        for (int j = 0; j < 4; ++j) {
            const int col = ((w & 1) * 4 + j) * 16 + lm;
            const float bn = in_b[col];
            #pragma unroll
            for (int r = 0; r < 8; ++r)
                Xf[(mt * 16 + r + 8 * g) * HID + col] = acc[j][r] + bn;
        }
    }
    __syncthreads();

    // ---- 4 blocks ----
    for (int blk = 0; blk < 4; ++blk) {
        layernorm_bf16(Xf, LNX, ln1g + blk * HID, ln1b + blk * HID, tid);
        __syncthreads();

        // depthwise conv over T (K=5, pad 2) + residual
        {
            const float* cw = convw + blk * 5 * HID;
            const float* cb = convb + blk * HID;
            for (int idx = tid; idx < TN * HID; idx += 256) {
                const int t = idx >> 7, d = idx & 127;
                float acc = cb[d];
                #pragma unroll
                for (int j = 0; j < 5; ++j) {
                    const int t2 = t + j - 2;
                    if (t2 >= 0 && t2 < TN) acc += cw[j * HID + d] * (float)LNX[t2 * HID + d];
                }
                Xf[idx] += acc;
            }
        }
        __syncthreads();

        layernorm_bf16(Xf, XB, ln2g + blk * HID, ln2b + blk * HID, tid);
        __syncthreads();

        // MLP1: [64,128]@[128,256] + bias + exact GELU -> HB  (8 N-tiles/wave)
        {
            f32x8 acc[8] = {};
            for (int kt = 0; kt < 4; ++kt) {
                FragBF a;
                const char* ab = (const char*)XB + (mt * 16 + lm) * 256 + kt * 64 + g * 16;
                a.q[0] = *(const uint4*)ab;
                a.q[1] = *(const uint4*)(ab + 32);
                #pragma unroll
                for (int j = 0; j < 8; ++j) {
                    const int n = ((w & 1) * 8 + j) * 16 + lm;
                    const __bf16* wp = W1T + (size_t)blk * 256 * 128 + n * 128 + kt * 32 + g * 8;
                    FragBF b;
                    b.q[0] = *(const uint4*)wp;
                    b.q[1] = *(const uint4*)(wp + 16);
                    acc[j] = __builtin_amdgcn_wmma_f32_16x16x32_bf16(
                        false, a.v, false, b.v, (short)0, acc[j], false, false);
                }
            }
            const float* bb = b1 + blk * 256;
            #pragma unroll
            for (int j = 0; j < 8; ++j) {
                const int col = ((w & 1) * 8 + j) * 16 + lm;
                const float bn = bb[col];
                #pragma unroll
                for (int r = 0; r < 8; ++r)
                    HB[(mt * 16 + r + 8 * g) * 256 + col] = (__bf16)gelu_exact(acc[j][r] + bn);
            }
        }
        __syncthreads();

        // MLP2: [64,256]@[256,128] + bias + residual into Xf (4 N-tiles/wave)
        {
            f32x8 acc[4] = {};
            for (int kt = 0; kt < 8; ++kt) {
                FragBF a;
                const char* ab = (const char*)HB + (mt * 16 + lm) * 512 + kt * 64 + g * 16;
                a.q[0] = *(const uint4*)ab;
                a.q[1] = *(const uint4*)(ab + 32);
                #pragma unroll
                for (int j = 0; j < 4; ++j) {
                    const int n = ((w & 1) * 4 + j) * 16 + lm;
                    const __bf16* wp = W2T + (size_t)blk * 128 * 256 + n * 256 + kt * 32 + g * 8;
                    FragBF b;
                    b.q[0] = *(const uint4*)wp;
                    b.q[1] = *(const uint4*)(wp + 16);
                    acc[j] = __builtin_amdgcn_wmma_f32_16x16x32_bf16(
                        false, a.v, false, b.v, (short)0, acc[j], false, false);
                }
            }
            const float* bb = b2 + blk * HID;
            #pragma unroll
            for (int j = 0; j < 4; ++j) {
                const int col = ((w & 1) * 4 + j) * 16 + lm;
                const float bn = bb[col];
                #pragma unroll
                for (int r = 0; r < 8; ++r)
                    Xf[(mt * 16 + r + 8 * g) * HID + col] += acc[j][r] + bn;
            }
        }
        __syncthreads();
    }

    // ---- persist hidden via async LDS->global stores; overlap with heads ----
    {
        char* gdst = (char*)(hid + (size_t)q * TN * HID);
        const unsigned xfbase = lds_off(Xf);
        for (int i = tid; i < TN * HID * 4 / 16; i += 256) {
            const unsigned loff = xfbase + i * 16;
            char* ga = gdst + i * 16;
            asm volatile("global_store_async_from_lds_b128 %0, %1, off"
                         :: "v"(ga), "v"(loff) : "memory");
        }
    }

    // ---- delta / vis heads (tiny); Xf only read, async stores in flight ----
    if (tid < TN) {
        const int t = tid;
        float dx = 0.0f, dy = 0.0f;
        for (int d = 0; d < HID; ++d) {
            const float xv = Xf[t * HID + d];
            dx += xv * dw[d * 2];
            dy += xv * dw[d * 2 + 1];
        }
        dx += db[0]; dy += db[1];
        const int ti = (q * TN + t) * 2;
        const float nx = tracks[ti] + dx, ny = tracks[ti + 1] + dy;
        tracks[ti] = nx; tracks[ti + 1] = ny;
        if (last) {
            out_tracks[ti] = nx; out_tracks[ti + 1] = ny;
            float v = vb2[0];
            for (int h = 0; h < 64; ++h) {
                float a = vb1[h];
                for (int d = 0; d < HID; ++d) a += Xf[t * HID + d] * vw1[d * 64 + h];
                v += gelu_exact(a) * vw2[h];
            }
            out_vis[q * TN + t] = v;
        }
    }

    asm volatile("s_wait_asynccnt 0x0" ::: "memory");
}

// ---------------------------------------------------------------------------
extern "C" void kernel_launch(void* const* d_in, const int* in_sizes, int n_in,
                              void* d_out, int out_size, void* d_ws, size_t ws_size,
                              hipStream_t stream)
{
    const float* coarse = (const float*)d_in[0];
    const float* qf     = (const float*)d_in[1];
    const float* fm     = (const float*)d_in[2];
    const float* in_w   = (const float*)d_in[3];
    const float* in_b   = (const float*)d_in[4];
    const float* ln1g   = (const float*)d_in[5];
    const float* ln1b   = (const float*)d_in[6];
    const float* convw  = (const float*)d_in[7];
    const float* convb  = (const float*)d_in[8];
    const float* ln2g   = (const float*)d_in[9];
    const float* ln2b   = (const float*)d_in[10];
    const float* w1     = (const float*)d_in[11];
    const float* b1     = (const float*)d_in[12];
    const float* w2     = (const float*)d_in[13];
    const float* b2     = (const float*)d_in[14];
    const float* dw     = (const float*)d_in[15];
    const float* db     = (const float*)d_in[16];
    const float* vw1    = (const float*)d_in[17];
    const float* vb1    = (const float*)d_in[18];
    const float* vw2    = (const float*)d_in[19];
    const float* vb2    = (const float*)d_in[20];
    (void)in_sizes; (void)n_in; (void)out_size; (void)ws_size;

    char* ws = (char*)d_ws;
    float*  tracks = (float*)(ws + WS_TRACKS);
    float*  hid    = (float*)(ws + WS_HID);
    __bf16* xin    = (__bf16*)(ws + WS_XIN);
    __bf16* inT    = (__bf16*)(ws + WS_INT);
    __bf16* W1T    = (__bf16*)(ws + WS_W1T);
    __bf16* W2T    = (__bf16*)(ws + WS_W2T);

    float* out_tracks = (float*)d_out;          // [1,128,64,2]
    float* out_vis    = out_tracks + 16384;     // [1,128,64]

    const int prep_total = 16384 + 8192 * 128 + 128 * 288 + 2 * 4 * 256 * 128;
    prep_kernel<<<(prep_total + 255) / 256, 256, 0, stream>>>(
        coarse, in_w, w1, w2, tracks, hid, inT, W1T, W2T);

    for (int it = 0; it < 4; ++it) {
        corr_kernel<<<1024, 256, 0, stream>>>(fm, qf, tracks, hid, xin);
        refine_kernel<<<QN, 256, 135168, stream>>>(
            xin, inT, W1T, W2T, in_b, ln1g, ln1b, convw, convb, ln2g, ln2b,
            b1, b2, dw, db, vw1, vb1, vw2, vb2,
            tracks, hid, out_tracks, out_vis, (it == 3) ? 1 : 0);
    }
}
